// point_cloud_decoder_65524021068166
// MI455X (gfx1250) — matrix operand order repaired and checked by reading, not compiled, not run on hardware
//
#include <hip/hip_runtime.h>

// ---------------------------------------------------------------------------
// CDNA5 / gfx1250 point-cloud transformer decoder.
// All dense GEMMs run on v_wmma_f32_16x16x32_bf16 (wave32, f32 accumulate).
// Each wave computes a 32x64 C tile -> 8 independent WMMAs per k-step
// (back-to-back matrix issue, 2x A/B fragment reuse vs 16x64 tiles).
// rbf@We is fused into the attention kernel (16 k-rows == one WMMA A tile),
// avoiding the 134MB "e" tensor. Attention softmax einsums (0.5 GF) on VALU.
// ---------------------------------------------------------------------------

typedef __bf16 bf16x8  __attribute__((ext_vector_type(8)));
typedef __bf16 bf16x16 __attribute__((ext_vector_type(16)));
typedef float  f32x8   __attribute__((ext_vector_type(8)));

static constexpr int NPTS = 16384;   // N
static constexpr int D    = 256;
static constexpr int HD   = 128;     // H*DH
static constexpr int KN   = 16;      // neighbors
static constexpr int NRP  = 64;      // padded RBF features (50 -> 64)
static constexpr int KAUG = 288;     // padded D+3 (259 -> 288)
static constexpr int LAY  = 4;
static constexpr int OUTC = 103;
static constexpr int OUTP = 128;     // padded output cols

// ---------------------------------------------------------------------------
// Weight transpose + f32->bf16 convert:  src[Kreal][Nreal] -> dst[Npad][Kpad]
// (column-major bf16 so each WMMA B-fragment is two contiguous 16B loads)
// ---------------------------------------------------------------------------
__global__ void tconv_kernel(const float* __restrict__ src, __bf16* __restrict__ dst,
                             int Kreal, int Nreal, int Kpad, int Npad) {
  int idx = blockIdx.x * 256 + threadIdx.x;
  if (idx >= Kpad * Npad) return;
  int n = idx / Kpad, k = idx % Kpad;
  float v = (k < Kreal && n < Nreal) ? src[(size_t)k * Nreal + n] : 0.0f;
  dst[idx] = (__bf16)v;
}

// Xaug[n][0..287]: col0=1, cols1..255=encoding[batch_idx[n]], 256..258=pos, rest 0
__global__ void build_xaug(const float* __restrict__ enc, const float* __restrict__ pos,
                           const int* __restrict__ bidx, __bf16* __restrict__ X) {
  size_t idx = (size_t)blockIdx.x * 256 + threadIdx.x;
  if (idx >= (size_t)NPTS * KAUG) return;
  int n = (int)(idx / KAUG), c = (int)(idx % KAUG);
  float v;
  if (c == 0)        v = 1.0f;
  else if (c < 256)  v = enc[(size_t)bidx[n] * 256 + c];
  else if (c < 259)  v = pos[(size_t)n * 3 + (c - 256)];
  else               v = 0.0f;
  X[idx] = (__bf16)v;
}

// rbfp[(n*16+k)][j<64]: gaussian RBF of neighbor distance, zero-padded past 50
__global__ void build_rbf(const float* __restrict__ pos, const int* __restrict__ nbrs,
                          __bf16* __restrict__ rbfp) {
  size_t idx = (size_t)blockIdx.x * 256 + threadIdx.x;
  if (idx >= (size_t)NPTS * KN * NRP) return;
  int j = (int)(idx & (NRP - 1));
  size_t nk = idx >> 6;
  if (j >= 50) { rbfp[idx] = (__bf16)0.0f; return; }
  int n = (int)(nk >> 4), k = (int)(nk & 15);
  int nb = nbrs[n * KN + k];
  float dx = pos[(size_t)nb * 3 + 0] - pos[(size_t)n * 3 + 0];
  float dy = pos[(size_t)nb * 3 + 1] - pos[(size_t)n * 3 + 1];
  float dz = pos[(size_t)nb * 3 + 2] - pos[(size_t)n * 3 + 2];
  float d = sqrtf(dx * dx + dy * dy + dz * dz + 1e-12f);
  const float w = 2.0f / 49.0f;
  float u = (d - (float)j * w) / w;
  rbfp[idx] = (__bf16)__expf(-0.5f * u * u);
}

// ---------------------------------------------------------------------------
// Fragment helpers (ISA 7.12.2 wave32 layouts)
// A 16x32 bf16: lane<16 -> row=lane, K{0..7,16..23}; lane>=16 -> K{8..15,24..31}
// B 32x16 bf16: lane&15 = column, lane<16 -> K{0..7,16..23}; lane>=16 -> K{8..15,24..31}
// C 16x16 f32 : vgpr r -> M = 8*(lane>>4)+r, N = lane&15
// ---------------------------------------------------------------------------
__device__ __forceinline__ bf16x16 load_a_frag(const __bf16* Arow, int k0, int half) {
  bf16x8 lo = *(const bf16x8*)(Arow + k0 + half * 8);
  bf16x8 hi = *(const bf16x8*)(Arow + k0 + 16 + half * 8);
  bf16x16 a;
  #pragma unroll
  for (int i = 0; i < 8; ++i) { a[i] = lo[i]; a[8 + i] = hi[i]; }
  return a;
}
__device__ __forceinline__ bf16x16 load_b_frag(const __bf16* Bcol, int k0, int half) {
  bf16x8 lo = *(const bf16x8*)(Bcol + k0 + half * 8);
  bf16x8 hi = *(const bf16x8*)(Bcol + k0 + 16 + half * 8);
  bf16x16 b;
  #pragma unroll
  for (int i = 0; i < 8; ++i) { b[i] = lo[i]; b[8 + i] = hi[i]; }
  return b;
}

__device__ __forceinline__ float gelu_tanh(float x) {
  return 0.5f * x * (1.0f + tanhf(0.7978845608028654f * (x + 0.044715f * x * x * x)));
}

// ---------------------------------------------------------------------------
// WMMA bf16 GEMM: C[M x ncols] = A[M x Ksz] @ Bt^T  (Bt stored [ncols_pad][Ksz])
// 128 threads = 4 waves; wave w computes rows [128*bx+32w, +32) x cols [64*by,+64)
// as 2x4 16x16 WMMA tiles: 8 independent WMMAs per 32-deep k-step.
// Epilogue: +bias, +residual, tanh-GELU, dual f32/bf16 writeback.
// ---------------------------------------------------------------------------
template <bool BIAS, bool RES, bool GELU_, bool WB16>
__global__ __launch_bounds__(128)
void gemm_bf16_wmma(const __bf16* __restrict__ A, const __bf16* __restrict__ Bt,
                    int Ksz, float* __restrict__ C32, __bf16* __restrict__ C16,
                    int ldc, int ncols_store, const float* __restrict__ bias,
                    const float* __restrict__ Rsd) {
  const int lane = threadIdx.x & 31;
  const int wave = threadIdx.x >> 5;
  const int half = lane >> 4;    // which half-wave (selects K sub-block)
  const int l16  = lane & 15;    // row (A) / col (B,C) within tile
  const int rowbase = blockIdx.x * 128 + wave * 32;
  const int colbase = blockIdx.y * 64;

  f32x8 acc[2][4];
  #pragma unroll
  for (int m = 0; m < 2; ++m)
    #pragma unroll
    for (int t = 0; t < 4; ++t) acc[m][t] = f32x8{};

  const __bf16* Arow0 = A + (size_t)(rowbase + l16) * Ksz;
  const __bf16* Arow1 = A + (size_t)(rowbase + 16 + l16) * Ksz;

  for (int k0 = 0; k0 < Ksz; k0 += 32) {
    __builtin_prefetch(Arow0 + k0 + 64, 0, 1);   // global_prefetch_b8
    __builtin_prefetch(Arow1 + k0 + 64, 0, 1);
    bf16x16 a0 = load_a_frag(Arow0, k0, half);
    bf16x16 a1 = load_a_frag(Arow1, k0, half);
    bf16x16 b[4];
    #pragma unroll
    for (int t = 0; t < 4; ++t)
      b[t] = load_b_frag(Bt + (size_t)(colbase + t * 16 + l16) * Ksz, k0, half);
    #pragma unroll
    for (int t = 0; t < 4; ++t) {
      acc[0][t] = __builtin_amdgcn_wmma_f32_16x16x32_bf16(
          false, a0, false, b[t], (short)0, acc[0][t], false, false);
      acc[1][t] = __builtin_amdgcn_wmma_f32_16x16x32_bf16(
          false, a1, false, b[t], (short)0, acc[1][t], false, false);
    }
  }

  #pragma unroll
  for (int t = 0; t < 4; ++t) {
    int col = colbase + t * 16 + l16;
    if (col >= ncols_store) continue;
    float bv = BIAS ? bias[col] : 0.0f;
    #pragma unroll
    for (int m = 0; m < 2; ++m) {
      #pragma unroll
      for (int r = 0; r < 8; ++r) {
        int row = rowbase + m * 16 + half * 8 + r;
        float v = acc[m][t][r] + bv;
        if (RES)   v += Rsd[(size_t)row * ldc + col];
        if (GELU_) v = gelu_tanh(v);
        C32[(size_t)row * ldc + col] = v;
        if (WB16) C16[(size_t)row * ldc + col] = (__bf16)v;
      }
    }
  }
}

// ---------------------------------------------------------------------------
// Attention: one block (128 thr = 4 waves) per node.
//  1. gather q, k[nbr], v[nbr] into LDS
//  2. e = rbf(16x64) @ WeT  via WMMA (wave w -> cols [32w,32w+32)), += kj,vj
//  3. logits (64 lanes = (k,h) pairs, dot over DH=32), masked softmax, msg
// ---------------------------------------------------------------------------
__global__ __launch_bounds__(128)
void attn_kernel(const float* __restrict__ Q, const float* __restrict__ Km,
                 const float* __restrict__ Vm, const __bf16* __restrict__ rbfp,
                 const __bf16* __restrict__ WeT, const int* __restrict__ nbrs,
                 const float* __restrict__ nmask, __bf16* __restrict__ msg16) {
  __shared__ float qs[HD];
  __shared__ float kjs[KN][HD];
  __shared__ float vjs[KN][HD];
  __shared__ float lg[4][KN];
  __shared__ float aw[4][KN];
  __shared__ float mk[KN];

  const int n = blockIdx.x;
  const int t = threadIdx.x;
  const int lane = t & 31, wave = t >> 5, half = lane >> 4, l16 = lane & 15;

  qs[t] = Q[(size_t)n * HD + t];
  if (t < KN) mk[t] = nmask[n * KN + t];
  for (int k = 0; k < KN; ++k) {
    int nb = nbrs[n * KN + k];
    kjs[k][t] = Km[(size_t)nb * HD + t];
    vjs[k][t] = Vm[(size_t)nb * HD + t];
  }
  __syncthreads();

  // e = rbf @ We  (A rows are the 16 neighbors -> exactly one WMMA A tile)
  const __bf16* Ar = rbfp + ((size_t)n * KN + l16) * NRP;
  #pragma unroll
  for (int tt = 0; tt < 2; ++tt) {
    int colb = wave * 32 + tt * 16;
    f32x8 acc = {};
    #pragma unroll
    for (int k0 = 0; k0 < NRP; k0 += 32) {
      bf16x16 a = load_a_frag(Ar, k0, half);
      bf16x16 b = load_b_frag(WeT + (size_t)(colb + l16) * NRP, k0, half);
      acc = __builtin_amdgcn_wmma_f32_16x16x32_bf16(
          false, a, false, b, (short)0, acc, false, false);
    }
    #pragma unroll
    for (int r = 0; r < 8; ++r) {
      int kk = half * 8 + r, cc = colb + l16;   // (neighbor, feature)
      kjs[kk][cc] += acc[r];
      vjs[kk][cc] += acc[r];
    }
  }
  __syncthreads();

  if (t < 64) {  // (k,h) lanes
    int k = t & 15, h = t >> 4;
    float s = 0.0f;
    #pragma unroll
    for (int d = 0; d < 32; ++d) s += qs[h * 32 + d] * kjs[k][h * 32 + d];
    s *= 0.17677669529663687f;  // 1/sqrt(32)
    lg[h][k] = (mk[k] > 0.0f) ? s : -1.0e9f;
  }
  __syncthreads();

  if (t < 4) {   // per-head softmax over 16 neighbors
    int h = t;
    float m = -1.0e30f;
    for (int k = 0; k < KN; ++k) m = fmaxf(m, lg[h][k]);
    float ex[KN], ssum = 0.0f;
    for (int k = 0; k < KN; ++k) { ex[k] = __expf(lg[h][k] - m); ssum += ex[k]; }
    float inv = 1.0f / ssum;
    for (int k = 0; k < KN; ++k) aw[h][k] = ex[k] * inv * mk[k];
  }
  __syncthreads();

  {  // msg: lane = (h,d)
    int h = t >> 5, d = t & 31;
    float s = 0.0f;
    #pragma unroll
    for (int k = 0; k < KN; ++k) s += aw[h][k] * vjs[k][h * 32 + d];
    msg16[(size_t)n * HD + t] = (__bf16)s;
  }
}

// ---------------------------------------------------------------------------
// Row LayerNorm (256 threads per row) -> bf16 output
// ---------------------------------------------------------------------------
__global__ __launch_bounds__(256)
void ln_kernel(const float* __restrict__ X, const float* __restrict__ g,
               const float* __restrict__ b, __bf16* __restrict__ out16) {
  __shared__ float red[256];
  int row = blockIdx.x, t = threadIdx.x;
  float v = X[(size_t)row * D + t];
  red[t] = v; __syncthreads();
  for (int s = 128; s > 0; s >>= 1) { if (t < s) red[t] += red[t + s]; __syncthreads(); }
  float mu = red[0] * (1.0f / 256.0f); __syncthreads();
  float dv = v - mu;
  red[t] = dv * dv; __syncthreads();
  for (int s = 128; s > 0; s >>= 1) { if (t < s) red[t] += red[t + s]; __syncthreads(); }
  float var = red[0] * (1.0f / 256.0f);
  out16[(size_t)row * D + t] = (__bf16)(g[t] * dv * rsqrtf(var + 1e-5f) + b[t]);
}

// ---------------------------------------------------------------------------
extern "C" void kernel_launch(void* const* d_in, const int* in_sizes, int n_in,
                              void* d_out, int out_size, void* d_ws, size_t ws_size,
                              hipStream_t stream) {
  const float* enc   = (const float*)d_in[0];
  const float* pos   = (const float*)d_in[1];
  const int*   bidx  = (const int*)  d_in[2];
  const int*   nbrs  = (const int*)  d_in[3];
  const float* nmask = (const float*)d_in[4];
  const float* W_in  = (const float*)d_in[5];
  const float* b_in  = (const float*)d_in[6];
  const float* Wq    = (const float*)d_in[7];
  const float* Wk    = (const float*)d_in[8];
  const float* Wv    = (const float*)d_in[9];
  const float* We    = (const float*)d_in[10];
  const float* Wo    = (const float*)d_in[11];
  const float* bo    = (const float*)d_in[12];
  const float* Wfc   = (const float*)d_in[13];
  const float* bfc   = (const float*)d_in[14];
  const float* ln_g  = (const float*)d_in[15];
  const float* ln_b  = (const float*)d_in[16];
  const float* W_out = (const float*)d_in[17];
  const float* b_out = (const float*)d_in[18];

  // ---- workspace carve (deterministic) ----
  char* base = (char*)d_ws;
  size_t off = 0;
  auto carve = [&](size_t bytes) { char* p = base + off; off = (off + bytes + 255) & ~(size_t)255; return p; };

  __bf16* WinT  = (__bf16*)carve((size_t)256 * KAUG * 2);
  __bf16* WqT   = (__bf16*)carve((size_t)LAY * HD * D * 2);
  __bf16* WkT   = (__bf16*)carve((size_t)LAY * HD * D * 2);
  __bf16* WvT   = (__bf16*)carve((size_t)LAY * HD * D * 2);
  __bf16* WeT   = (__bf16*)carve((size_t)LAY * HD * NRP * 2);
  __bf16* WoT   = (__bf16*)carve((size_t)LAY * D * HD * 2);
  __bf16* WfcT  = (__bf16*)carve((size_t)LAY * D * D * 2);
  __bf16* WoutT = (__bf16*)carve((size_t)OUTP * D * 2);
  __bf16* Xaug  = (__bf16*)carve((size_t)NPTS * KAUG * 2);
  float*  h32   = (float*) carve((size_t)NPTS * D * 4);
  __bf16* hA16  = (__bf16*)carve((size_t)NPTS * D * 2);
  __bf16* hB16  = (__bf16*)carve((size_t)NPTS * D * 2);
  __bf16* ln16  = (__bf16*)carve((size_t)NPTS * D * 2);
  float*  Qb    = (float*) carve((size_t)NPTS * HD * 4);
  float*  Kb    = (float*) carve((size_t)NPTS * HD * 4);
  float*  Vb    = (float*) carve((size_t)NPTS * HD * 4);
  __bf16* rbfp  = (__bf16*)carve((size_t)NPTS * KN * NRP * 2);
  __bf16* msg16 = (__bf16*)carve((size_t)NPTS * HD * 2);
  float*  t32   = (float*) carve((size_t)NPTS * D * 4);

  auto tconv = [&](const float* s, __bf16* dst, int Kr, int Nr, int Kp, int Np) {
    int tot = Kp * Np;
    tconv_kernel<<<(tot + 255) / 256, 256, 0, stream>>>(s, dst, Kr, Nr, Kp, Np);
  };

  // ---- weight prep ----
  tconv(W_in, WinT, 259, 256, KAUG, 256);
  for (int l = 0; l < LAY; ++l) {
    tconv(Wq + (size_t)l * D * HD, WqT + (size_t)l * HD * D, D, HD, D, HD);
    tconv(Wk + (size_t)l * D * HD, WkT + (size_t)l * HD * D, D, HD, D, HD);
    tconv(Wv + (size_t)l * D * HD, WvT + (size_t)l * HD * D, D, HD, D, HD);
    tconv(We + (size_t)l * 50 * HD, WeT + (size_t)l * HD * NRP, 50, HD, NRP, HD);
    tconv(Wo + (size_t)l * HD * D, WoT + (size_t)l * D * HD, HD, D, HD, D);
    tconv(Wfc + (size_t)l * D * D, WfcT + (size_t)l * D * D, D, D, D, D);
  }
  tconv(W_out, WoutT, D, OUTC, D, OUTP);

  // ---- activation prep ----
  {
    size_t tot = (size_t)NPTS * KAUG;
    build_xaug<<<(unsigned)((tot + 255) / 256), 256, 0, stream>>>(enc, pos, bidx, Xaug);
  }
  {
    size_t tot = (size_t)NPTS * KN * NRP;
    build_rbf<<<(unsigned)((tot + 255) / 256), 256, 0, stream>>>(pos, nbrs, rbfp);
  }

  const dim3 blk(128);
  const int MB = NPTS / 128;  // 128 row-blocks (32 rows per wave)

  // h = Xaug @ W_in + b_in   (f32 + bf16)
  gemm_bf16_wmma<true, false, false, true><<<dim3(MB, D / 64), blk, 0, stream>>>(
      Xaug, WinT, KAUG, h32, hA16, D, D, b_in, nullptr);

  __bf16* cur16 = hA16;
  __bf16* alt16 = hB16;

  for (int l = 0; l < LAY; ++l) {
    const __bf16* wq = WqT + (size_t)l * HD * D;
    const __bf16* wk = WkT + (size_t)l * HD * D;
    const __bf16* wv = WvT + (size_t)l * HD * D;
    const __bf16* we = WeT + (size_t)l * HD * NRP;
    const __bf16* wo = WoT + (size_t)l * D * HD;
    const __bf16* wf = WfcT + (size_t)l * D * D;

    // Q,K,V = h @ W{q,k,v}
    gemm_bf16_wmma<false, false, false, false><<<dim3(MB, HD / 64), blk, 0, stream>>>(
        cur16, wq, D, Qb, nullptr, HD, HD, nullptr, nullptr);
    gemm_bf16_wmma<false, false, false, false><<<dim3(MB, HD / 64), blk, 0, stream>>>(
        cur16, wk, D, Kb, nullptr, HD, HD, nullptr, nullptr);
    gemm_bf16_wmma<false, false, false, false><<<dim3(MB, HD / 64), blk, 0, stream>>>(
        cur16, wv, D, Vb, nullptr, HD, HD, nullptr, nullptr);

    // neighbor attention (rbf@We fused via WMMA)
    attn_kernel<<<NPTS, 128, 0, stream>>>(Qb, Kb, Vb, rbfp, we, nbrs, nmask, msg16);

    // t = h + msg @ Wo + bo
    gemm_bf16_wmma<true, true, false, false><<<dim3(MB, D / 64), blk, 0, stream>>>(
        msg16, wo, HD, t32, nullptr, D, D, bo + (size_t)l * D, h32);

    // LayerNorm -> bf16
    ln_kernel<<<NPTS, 256, 0, stream>>>(t32, ln_g + (size_t)l * D, ln_b + (size_t)l * D, ln16);

    // h = gelu(ln @ Wfc + bfc)  (f32 + bf16 into ping-pong buffer)
    gemm_bf16_wmma<true, false, true, true><<<dim3(MB, D / 64), blk, 0, stream>>>(
        ln16, wf, D, h32, alt16, D, D, bfc + (size_t)l * D, nullptr);

    __bf16* tmp = cur16; cur16 = alt16; alt16 = tmp;
  }

  // out = h @ W_out + b_out   (store only the real 103 columns)
  gemm_bf16_wmma<true, false, false, false><<<dim3(MB, OUTP / 64), blk, 0, stream>>>(
      cur16, WoutT, D, (float*)d_out, nullptr, OUTC, OUTC, b_out, nullptr);
}
